// SigModule_1614907703712
// MI455X (gfx1250) — matrix operand order repaired and checked by reading, not compiled, not run on hardware
//
#include <hip/hip_runtime.h>

typedef __attribute__((ext_vector_type(16))) _Float16 v16h;
typedef __attribute__((ext_vector_type(8)))  _Float16 v8h;
typedef __attribute__((ext_vector_type(8)))  float    v8f;

#define NB    64
#define CB    16
#define JB    25
#define TB    120
#define CCH   17          // C + 1 channels
#define SIGCH 306         // 17 + 17*17
#define KPAD  320         // K padded to multiple of 32
#define YSTR  328         // yh row stride in halves (conflict-free, 16B aligned)
#define PSTR  12          // plds k-stride in floats (48B: b128-aligned, bank-spread)
#define ODIM  64
#define RPB   64          // rows per block (4 WMMA M-tiles per wave)
#define NROWS (NB*JB*TB)  // 192000
#define NKF   (KPAD/32)   // 10 k-fragments
#define NPAIR 136         // C(17,2) unordered channel pairs
#define TASKS 184         // per row: 17 S1 + 17 diag + 136 pairs + 14 zero-pad
#define WFRAG_HALVES (4*NKF*32*16)  // 20480 halves = 40KB

// ---------------------------------------------------------------------------
// Kernel 0: pre-swizzle W (f32, 306x64) into f16 WMMA B-fragments in workspace.
// Layout: wsB[wtile][kf][lane][i]; element i of lane l = W[kf*32 + (i<8 ? i : i+8) + (l>=16?8:0)][wtile*16 + (l&15)]
// ---------------------------------------------------------------------------
__global__ void build_wfrag(const float* __restrict__ W, _Float16* __restrict__ wsB) {
  int id = blockIdx.x * 256 + threadIdx.x;
  if (id >= WFRAG_HALVES) return;
  int i  = id & 15;
  int l  = (id >> 4) & 31;
  int kf = (id >> 9) % NKF;
  int w  = id / (NKF * 512);
  int k  = kf * 32 + ((i < 8) ? i : i + 8) + ((l >= 16) ? 8 : 0);
  int n  = w * 16 + (l & 15);
  float v = (k < SIGCH) ? W[k * ODIM + n] : 0.0f;
  wsB[id] = (_Float16)v;
}

// A fragment from LDS: interleaved K pattern per ISA 16-bit A-matrix table.
__device__ __forceinline__ v16h load_afrag(const _Float16* p) {
  v8h lo = *(const v8h*)p;          // K = k0+base   .. +7
  v8h hi = *(const v8h*)(p + 16);   // K = k0+base+16 .. +23
  return __builtin_shufflevector(lo, hi,
      0, 1, 2, 3, 4, 5, 6, 7, 8, 9, 10, 11, 12, 13, 14, 15);
}

// ---------------------------------------------------------------------------
// Main kernel: one block = 64 signature rows; wave w owns columns [16w,16w+16)
// across all 64 rows (4 M-tiles), so each B fragment is register-reused 4x.
// ---------------------------------------------------------------------------
__global__ __launch_bounds__(128) void sig_wmma_kernel(
    const float* __restrict__ x, const _Float16* __restrict__ wsB,
    const float* __restrict__ bias, float* __restrict__ out) {

  __shared__ __align__(16) float plds[RPB][CCH][PSTR];   // path points, k-major
  __shared__ __align__(16) _Float16 yh[RPB * YSTR];      // y in f16 (A operand)

  const int tid  = threadIdx.x;
  const int row0 = blockIdx.x * RPB;

  // ---- Phase 1: windowed path points (x channels + time channel) ----------
  for (int idx = tid; idx < RPB * 5 * CCH; idx += 128) {
    int r   = idx / (5 * CCH);
    int rem = idx - r * (5 * CCH);
    int k   = rem / CCH;
    int ch  = rem - k * CCH;
    int R  = row0 + r;
    int n  = R / (JB * TB);
    int rj = R - n * (JB * TB);
    int j  = rj / TB;
    int t  = rj - j * TB;
    int pos = t - 2 + k;
    int pc  = pos < 0 ? 0 : (pos > TB - 1 ? TB - 1 : pos);
    float v;
    if (ch < CB) {
      v = x[((n * CB + ch) * JB + j) * TB + pc];
    } else {
      int start = (t - 2 < 0) ? 0 : t - 2;
      int end   = (t + 3 > TB) ? TB : t + 3;
      float L = (float)(end - start);
      v = (float)(pc - start) / (L - 1.0f);
    }
    plds[r][ch][k] = v;
  }
  __syncthreads();

  // ---- Phase 2: depth-2 signature -> y (f16) ------------------------------
  // S2 = 0.5*p4 (x) p4  +  antisymmetric Levy area:
  //   S2[a][b] = 0.5*p4[a]*p4[b] + 0.5*sum_{k=1..4}(p_{k-1}[a]p_k[b] - p_k[a]p_{k-1}[b])
  // Only 136 unordered pairs computed; each yields both (a,b) and (b,a).
  for (int idx = tid; idx < RPB * TASKS; idx += 128) {
    int r = idx / TASKS;
    int s = idx - r * TASKS;
    const float* prow = &plds[r][0][0];
    _Float16* yrow = &yh[r * YSTR];
    if (s < 17) {                       // S1 = last path point
      yrow[s] = (_Float16)prow[s * PSTR + 4];
    } else if (s < 34) {                // diagonal: 0.5 * p4[a]^2
      int a = s - 17;
      float p4 = prow[a * PSTR + 4];
      yrow[CCH + 17 * a + a] = (_Float16)(0.5f * p4 * p4);
    } else if (s < 34 + NPAIR) {        // off-diagonal pair
      int q = s - 34;
      int u = q / 17;                   // circular distance - 1 (0..7)
      int v = q - u * 17;               // first endpoint (0..16)
      int b2 = v + u + 1;
      if (b2 >= 17) b2 -= 17;
      int a = v < b2 ? v : b2;
      int b = v < b2 ? b2 : v;
      float4 pa03 = *(const float4*)(prow + a * PSTR);
      float4 pb03 = *(const float4*)(prow + b * PSTR);
      float  pa4  = prow[a * PSTR + 4];
      float  pb4  = prow[b * PSTR + 4];
      float area = pa03.x * pb03.y - pa03.y * pb03.x
                 + pa03.y * pb03.z - pa03.z * pb03.y
                 + pa03.z * pb03.w - pa03.w * pb03.z
                 + pa03.w * pb4    - pa4    * pb03.w;
      float prod = 0.5f * pa4 * pb4;
      float ha   = 0.5f * area;
      yrow[CCH + 17 * a + b] = (_Float16)(prod + ha);
      yrow[CCH + 17 * b + a] = (_Float16)(prod - ha);
    } else {                            // zero-pad K in [306, 320)
      yrow[SIGCH + (s - (34 + NPAIR))] = (_Float16)0.0f;
    }
  }
  __syncthreads();

  // ---- Phase 3: WMMA GEMM y[64 x 320] @ Wfrag[320 x 16] per wave ----------
  const int lane = tid & 31;
  const int w    = tid >> 5;        // wave -> column tile
  const int hi   = lane >> 4;       // upper lane half
  const int rsel = lane & 15;

  v8f acc[4] = {};
  const _Float16* wbase = wsB + (w * NKF * 32 + lane) * 16;
  const _Float16* abase = &yh[rsel * YSTR + (hi ? 8 : 0)];

  #pragma unroll
  for (int kf = 0; kf < NKF; ++kf) {
    v16h bfrag = *(const v16h*)(wbase + kf * 512);   // loaded once, reused 4x
    #pragma unroll
    for (int rt = 0; rt < 4; ++rt) {
      v16h afrag = load_afrag(abase + rt * 16 * YSTR + kf * 32);
      acc[rt] = __builtin_amdgcn_wmma_f32_16x16x32_f16(
          /*neg_a=*/false, afrag, /*neg_b=*/false, bfrag,
          /*c_mod=*/(short)0, acc[rt], /*reuse_a=*/false, /*reuse_b=*/false);
    }
  }

  // ---- Bias + vectorized transposed store to (N, 64, J, T) ----------------
  // Each lane's 8 accumulator rows are 8 consecutive t in one (n,j) run
  // (group start multiple of 8, 120 % 8 == 0) -> two aligned b128 stores.
  const int o  = w * 16 + rsel;
  const float bv = bias[o];
  #pragma unroll
  for (int rt = 0; rt < 4; ++rt) {
    int R0 = row0 + rt * 16 + (hi ? 8 : 0);   // first of 8 consecutive rows
    int n  = R0 / (JB * TB);
    int rj = R0 - n * (JB * TB);
    int j  = rj / TB;
    int t  = rj - j * TB;
    float* p = &out[((n * ODIM + o) * JB + j) * TB + t];
    float4 v0 = make_float4(acc[rt][0] + bv, acc[rt][1] + bv,
                            acc[rt][2] + bv, acc[rt][3] + bv);
    float4 v1 = make_float4(acc[rt][4] + bv, acc[rt][5] + bv,
                            acc[rt][6] + bv, acc[rt][7] + bv);
    *(float4*)(p)     = v0;
    *(float4*)(p + 4) = v1;
  }
}

// ---------------------------------------------------------------------------
extern "C" void kernel_launch(void* const* d_in, const int* in_sizes, int n_in,
                              void* d_out, int out_size, void* d_ws, size_t ws_size,
                              hipStream_t stream) {
  const float* x    = (const float*)d_in[0];   // (64,16,25,120) f32
  const float* W    = (const float*)d_in[1];   // (306,64) f32
  const float* bias = (const float*)d_in[2];   // (64,) f32
  float*       out  = (float*)d_out;           // (64,64,25,120) f32
  _Float16*    wsB  = (_Float16*)d_ws;         // 40KB of pre-swizzled B fragments

  build_wfrag<<<(WFRAG_HALVES + 255) / 256, 256, 0, stream>>>(W, wsB);
  sig_wmma_kernel<<<NROWS / RPB, 128, 0, stream>>>(x, wsB, bias, out);
}
